// IndGRU_81286551044245
// MI455X (gfx1250) — matrix operand order repaired and checked by reading, not compiled, not run on hardware
//
#include <hip/hip_runtime.h>
#include <hip/hip_bf16.h>

// ---------------------------------------------------------------------------
// IndGRU for MI455X (gfx1250, wave32, WMMA).
//   Phase 0: split W_ih once into bf16 hi/lo arrays (removes per-wave cvt
//            VALU work + WMMA->VALU WAR hazard NOPs from the GEMM loop).
//   Phase 1: gx[T*B, 3H] = x[T*B, I] @ W_ih^T  (+ b_ih), via
//            v_wmma_f32_16x16x32_bf16 with bf16 hi/lo split-precision
//            (hi*hi + hi*lo + lo*hi ~= 16 mantissa bits, near-fp32).
//   Phase 2: diagonal-recurrence scan, 16384 independent chains, with
//            register double-buffering + global_prefetch_b8 look-ahead.
// ---------------------------------------------------------------------------

typedef __attribute__((ext_vector_type(16))) __bf16 v16bf;
typedef __attribute__((ext_vector_type(8)))  float  v8f;

#define T_STEPS 2048
#define BSZ     64
#define ISZ     256
#define HSZ     256
#define N3H     768            // 3*H
#define MROWS   (T_STEPS * BSZ)
#define WELTS   (N3H * ISZ)    // 196608 elements of W_ih

// ---- Phase 0: one-time bf16 hi/lo split of W_ih ---------------------------
__global__ __launch_bounds__(256)
void w_split_kernel(const float* __restrict__ W,
                    __bf16* __restrict__ whi,
                    __bf16* __restrict__ wlo) {
  const int i = blockIdx.x * 256 + threadIdx.x;
  if (i < WELTS) {
    const float w = W[i];
    const __bf16 h = (__bf16)w;
    whi[i] = h;
    wlo[i] = (__bf16)(w - (float)h);
  }
}

// ---- bf16 split of the A operand (x rows), done in-registers --------------
// A-matrix 16-bit per-lane k order used by this code (contiguous, same as B):
//   lanes 0-15 : row M=lane,    elem j -> K = j
//   lanes 16-31: row M=lane-16, elem j -> K = 16 + j
// => two contiguous groups of 8 floats at (kc + off) and (kc + 8 + off),
//    off = 16*(lane>=16).
__device__ __forceinline__ void splitA(const float* __restrict__ base, int off,
                                       v16bf& hi, v16bf& lo) {
  const float4* p0 = reinterpret_cast<const float4*>(base + off);
  const float4* p1 = reinterpret_cast<const float4*>(base + 8 + off);
  float f[16];
  float4 q;
  q = p0[0]; f[0]  = q.x; f[1]  = q.y; f[2]  = q.z; f[3]  = q.w;
  q = p0[1]; f[4]  = q.x; f[5]  = q.y; f[6]  = q.z; f[7]  = q.w;
  q = p1[0]; f[8]  = q.x; f[9]  = q.y; f[10] = q.z; f[11] = q.w;
  q = p1[1]; f[12] = q.x; f[13] = q.y; f[14] = q.z; f[15] = q.w;
#pragma unroll
  for (int j = 0; j < 16; ++j) {
    __bf16 h = (__bf16)f[j];
    hi[j] = h;
    lo[j] = (__bf16)(f[j] - (float)h);
  }
}

// ---- Phase 1: gx = x @ W^T + b_ih -----------------------------------------
// grid (MROWS/128, N3H/64), block 256 (= 8 waves).
// Each wave: one 16-row M tile x 64-col N strip (4 C accumulators),
// A regs reused across the 4 N tiles; 3 WMMAs per (tile, K-chunk).
// B-matrix 16-bit per-lane k order used by this code (contiguous, same as A):
//   lanes 0-15 : col N=lane,    elem j -> K = j       (contig 16 @ kc)
//   lanes 16-31: col N=lane-16, elem j -> K = 16 + j  (contig 16 @ kc+16)
// Since B[k,n] = W_ih[n,k], a lane's 16 B elements are 32 contiguous bytes
// of the pre-split bf16 W arrays -> direct v16bf (2x b128) loads, no cvt.
__global__ __launch_bounds__(256)
void indgru_gemm_kernel(const float* __restrict__ x,
                        const __bf16* __restrict__ whi,
                        const __bf16* __restrict__ wlo,
                        const float* __restrict__ b_ih,
                        float* __restrict__ gx) {
  const int wave   = threadIdx.x >> 5;
  const int lane   = threadIdx.x & 31;
  const int lane16 = lane & 15;
  const int hhalf  = lane >> 4;
  const int aoff   = hhalf * 16;

  const int m0     = (blockIdx.x * 8 + wave) * 16;   // M tile base row
  const int n_base = blockIdx.y * 64;                // N strip base col

  const float* xrow = x + (size_t)(m0 + lane16) * ISZ;

  v8f acc[4];
#pragma unroll
  for (int nt = 0; nt < 4; ++nt)
    acc[nt] = v8f{0.f, 0.f, 0.f, 0.f, 0.f, 0.f, 0.f, 0.f};

  for (int kc = 0; kc < ISZ; kc += 32) {
    v16bf ahi, alo;
    splitA(xrow + kc, aoff, ahi, alo);
#pragma unroll
    for (int nt = 0; nt < 4; ++nt) {
      const size_t woff =
          (size_t)(n_base + nt * 16 + lane16) * ISZ + kc + hhalf * 16;
      const v16bf bhi = *reinterpret_cast<const v16bf*>(whi + woff);
      const v16bf blo = *reinterpret_cast<const v16bf*>(wlo + woff);
      acc[nt] = __builtin_amdgcn_wmma_f32_16x16x32_bf16(
          false, ahi, false, bhi, (short)0, acc[nt], false, false);
      acc[nt] = __builtin_amdgcn_wmma_f32_16x16x32_bf16(
          false, ahi, false, blo, (short)0, acc[nt], false, false);
      acc[nt] = __builtin_amdgcn_wmma_f32_16x16x32_bf16(
          false, alo, false, bhi, (short)0, acc[nt], false, false);
    }
  }

  // C/D layout: VGPR r -> (M = r + 8*hhalf, N = lane16) per ISA 7.12.2.
#pragma unroll
  for (int nt = 0; nt < 4; ++nt) {
    const int n = n_base + nt * 16 + lane16;
    const float bias = b_ih[n];
#pragma unroll
    for (int r = 0; r < 8; ++r) {
      const int m = m0 + r + hhalf * 8;
      gx[(size_t)m * N3H + n] = acc[nt][r] + bias;
    }
  }
}

// ---- Phase 2: diagonal-recurrence scan ------------------------------------
__device__ __forceinline__ float fsigmoid(float v) {
  return 1.0f / (1.0f + __expf(-v));
}
__device__ __forceinline__ float ftanh(float v) {
  return 1.0f - 2.0f / (__expf(2.0f * v) + 1.0f);
}

__global__ __launch_bounds__(128)
void indgru_scan_kernel(const float* __restrict__ gx,
                        const float* __restrict__ w_hh,
                        const float* __restrict__ b_hh,
                        float* __restrict__ out,
                        float* __restrict__ h_n) {
  const int tid = blockIdx.x * blockDim.x + threadIdx.x;   // 0..16383
  const int b = tid >> 8;
  const int h = tid & (HSZ - 1);

  const float w_hr = w_hh[h], w_hz = w_hh[HSZ + h], w_hn = w_hh[2 * HSZ + h];
  const float b_hr = b_hh[h], b_hz = b_hh[HSZ + h], b_hn = b_hh[2 * HSZ + h];

  const float* g0 = gx + (size_t)b * N3H + h;
  float* op = out + (size_t)b * HSZ + h;
  constexpr size_t GSTRIDE = (size_t)BSZ * N3H;   // gx stride per t
  constexpr size_t OSTRIDE = (size_t)BSZ * HSZ;   // out stride per t
  constexpr int TB = 4;
  constexpr int PF = 64;                          // prefetch look-ahead

  float ar[TB], az[TB], an[TB];   // buffer A
  float br[TB], bz[TB], bn[TB];   // buffer B
  float hc = 0.0f;

  auto loadb = [&](int t, float* r_, float* z_, float* n_) {
#pragma unroll
    for (int i = 0; i < TB; ++i) {
      const float* p = g0 + (size_t)(t + i) * GSTRIDE;
      r_[i] = p[0];
      z_[i] = p[HSZ];
      n_[i] = p[2 * HSZ];
    }
  };
  auto prefetchb = [&](int t) {
    if (t + TB <= T_STEPS) {
#pragma unroll
      for (int i = 0; i < TB; ++i) {
        const float* p = g0 + (size_t)(t + i) * GSTRIDE;
        __builtin_prefetch(p, 0, 3);
        __builtin_prefetch(p + HSZ, 0, 3);
        __builtin_prefetch(p + 2 * HSZ, 0, 3);
      }
    }
  };
  auto compute = [&](int t, const float* r_, const float* z_, const float* n_) {
#pragma unroll
    for (int i = 0; i < TB; ++i) {
      const float r = fsigmoid(r_[i] + w_hr * hc + b_hr);
      const float z = fsigmoid(z_[i] + w_hz * hc + b_hz);
      const float n = ftanh(n_[i] + r * (w_hn * hc + b_hn));
      hc = (1.0f - z) * n + z * hc;
      op[(size_t)(t + i) * OSTRIDE] = hc;
    }
  };

  loadb(0, ar, az, an);
  for (int t0 = 0; t0 < T_STEPS; t0 += 2 * TB) {
    loadb(t0 + TB, br, bz, bn);          // always in range (T % 8 == 0)
    prefetchb(t0 + PF);
    compute(t0, ar, az, an);
    if (t0 + 2 * TB < T_STEPS) loadb(t0 + 2 * TB, ar, az, an);
    prefetchb(t0 + PF + TB);
    compute(t0 + TB, br, bz, bn);
  }
  h_n[tid] = hc;
}

// ---------------------------------------------------------------------------
extern "C" void kernel_launch(void* const* d_in, const int* in_sizes, int n_in,
                              void* d_out, int out_size, void* d_ws,
                              size_t ws_size, hipStream_t stream) {
  (void)in_sizes; (void)n_in; (void)out_size; (void)ws_size;
  const float* x    = (const float*)d_in[0];  // [T, B, I]
  const float* W_ih = (const float*)d_in[1];  // [3H, I]
  const float* b_ih = (const float*)d_in[2];  // [3H]
  const float* b_hh = (const float*)d_in[3];  // [3H]
  const float* w_hh = (const float*)d_in[4];  // [3, H]

  float* out = (float*)d_out;                         // [T, B, H]
  float* h_n = out + (size_t)T_STEPS * BSZ * HSZ;     // [1, B, H]

  // Workspace layout: gx fp32 [T*B, 3H] | W_hi bf16 [3H*I] | W_lo bf16 [3H*I]
  char* ws = (char*)d_ws;
  float* gx   = (float*)ws;
  __bf16* whi = (__bf16*)(ws + (size_t)MROWS * N3H * sizeof(float));
  __bf16* wlo = whi + WELTS;

  w_split_kernel<<<dim3((WELTS + 255) / 256), dim3(256), 0, stream>>>(
      W_ih, whi, wlo);

  dim3 ggrid(MROWS / 128, N3H / 64);   // (1024, 12)
  indgru_gemm_kernel<<<ggrid, dim3(256), 0, stream>>>(x, whi, wlo, b_ih, gx);

  indgru_scan_kernel<<<dim3((BSZ * HSZ) / 128), dim3(128), 0, stream>>>(
      gx, w_hh, b_hh, out, h_n);
}
